// Turbulence_57080115363915
// MI455X (gfx1250) — matrix operand order repaired and checked by reading, not compiled, not run on hardware
//
#include <hip/hip_runtime.h>
#include <hip/hip_bf16.h>
#include <math.h>

// ---------------------------------------------------------------------------
// Problem constants (from reference): B=32, C=8, S=512, D=256, stack Sg=4
// ---------------------------------------------------------------------------
#define BB 32
#define CC 8
#define SS 512
#define DD 256

typedef _Float16 h16;
typedef __attribute__((ext_vector_type(16))) _Float16 v16h;
typedef __attribute__((ext_vector_type(8)))  _Float16 v8h;
typedef __attribute__((ext_vector_type(8)))  float    v8f;

// ----------------------------- complex helpers -----------------------------
__device__ __forceinline__ float2 cmul(float2 a, float2 b) {
    return make_float2(a.x * b.x - a.y * b.y, a.x * b.y + a.y * b.x);
}
__device__ __forceinline__ float2 cadd(float2 a, float2 b) {
    return make_float2(a.x + b.x, a.y + b.y);
}
// complex sin: sin(a+bi) = sin a cosh b + i cos a sinh b
__device__ __forceinline__ float2 csinc(float2 z) {
    float sa, ca;
    __sincosf(z.x, &sa, &ca);
    float e  = __expf(z.y);
    float ei = 1.0f / e;
    float ch = 0.5f * (e + ei);
    float sh = 0.5f * (e - ei);
    return make_float2(sa * ch, ca * sh);
}

// ---------------------------------------------------------------------------
// 512-point radix-2 DIT FFT in LDS. One block (256 threads) per vector.
// dirSign = -1 : forward fft (exp(-2*pi*i))     scale = 1
// dirSign = +1 : inverse fft (exp(+2*pi*i))     scale = 1/512
// realInput: in is const float* (imag = 0), else const float2*
// ---------------------------------------------------------------------------
__global__ __launch_bounds__(256) void fft512_kernel(float2* __restrict__ out,
                                                     const void* __restrict__ in,
                                                     float dirSign, float scale,
                                                     int realInput) {
    __shared__ float sr[SS];
    __shared__ float si[SS];
    const int vec = blockIdx.x;
    const int tid = threadIdx.x;
    const float2* cin = (const float2*)in;
    const float*  rin = (const float*)in;

    for (int r = 0; r < 2; ++r) {
        int idx = tid + r * 256;
        int rev = (int)(__brev((unsigned)idx) >> 23);  // 9-bit reversal
        float re, im;
        if (realInput) { re = rin[(size_t)vec * SS + idx]; im = 0.0f; }
        else { float2 v = cin[(size_t)vec * SS + idx]; re = v.x; im = v.y; }
        sr[rev] = re;
        si[rev] = im;
    }
    __syncthreads();

    for (int st = 1; st <= 9; ++st) {
        int m  = 1 << st;
        int mh = m >> 1;
        int grp = tid / mh;
        int pos = tid - grp * mh;
        int i0 = grp * m + pos;
        int i1 = i0 + mh;
        float ang = dirSign * 6.28318530717958647692f * (float)pos / (float)m;
        float ws, wc;
        __sincosf(ang, &ws, &wc);
        float tr = wc * sr[i1] - ws * si[i1];
        float ti = wc * si[i1] + ws * sr[i1];
        sr[i1] = sr[i0] - tr;
        si[i1] = si[i0] - ti;
        sr[i0] += tr;
        si[i0] += ti;
        __syncthreads();
    }

    for (int r = 0; r < 2; ++r) {
        int idx = tid + r * 256;
        out[(size_t)vec * SS + idx] = make_float2(sr[idx] * scale, si[idx] * scale);
    }
}

// ---------------------------------------------------------------------------
// Knot: out[b,k,s] = r[k] + mean_d( w[k,d] * sin(f[k,d]*xe[b,s] + p[k,d]) )
// If Kin != Kk, xe = mean over Kin (collapse), else xe = x[b,k,s].
// ---------------------------------------------------------------------------
__global__ __launch_bounds__(256) void knot_kernel(const void* __restrict__ x,
                                                   int realInput, int Kin,
                                                   const float2* __restrict__ w,
                                                   const float2* __restrict__ f,
                                                   const float2* __restrict__ p,
                                                   const float2* __restrict__ r,
                                                   int Kk,
                                                   float2* __restrict__ out,
                                                   long outOffset, long outStrideB) {
    long gid = (long)blockIdx.x * blockDim.x + threadIdx.x;
    int  s = (int)(gid % SS);
    int  k = (int)((gid / SS) % Kk);
    int  b = (int)(gid / ((long)SS * Kk));
    const float2* cx = (const float2*)x;
    const float*  rx = (const float*)x;

    float2 xe;
    if (Kin == Kk) {
        long xi = ((long)b * Kin + k) * SS + s;
        xe = realInput ? make_float2(rx[xi], 0.0f) : cx[xi];
    } else {  // collapse: mean over Kin, broadcast over k
        float2 acc = make_float2(0.0f, 0.0f);
        for (int kk = 0; kk < Kin; ++kk) {
            long xi = ((long)b * Kin + kk) * SS + s;
            float2 v = realInput ? make_float2(rx[xi], 0.0f) : cx[xi];
            acc = cadd(acc, v);
        }
        xe = make_float2(acc.x / Kin, acc.y / Kin);
    }

    float2 acc = make_float2(0.0f, 0.0f);
    const long kb = (long)k * DD;
    for (int d = 0; d < DD; ++d) {
        float2 fv = f[kb + d];
        float2 pv = p[kb + d];
        float2 wv = w[kb + d];
        float2 sz = csinc(cadd(cmul(fv, xe), pv));
        acc = cadd(acc, cmul(wv, sz));
    }
    float2 res = cadd(r[k], make_float2(acc.x / DD, acc.y / DD));
    out[outOffset + (long)b * outStrideB + (long)k * SS + s] = res;
}

// ---------------------------------------------------------------------------
// Pack fft'd activations into WMMA-A fragment order (global memory):
//   Afrag[c][kblk][mTile][lane][e']  (e' 0..15 = re, 16..31 = im)
// WMMA 16-bit A 16x32 layout: lane = (m&15) + 16*hf,
//   k(e,hf) = e + 8*hf + (e>=8 ? 8 : 0)
// Source element: xf[b*bStride + j*jStrideA + c*cStride + t], j=k>>9, t=k&511
// ---------------------------------------------------------------------------
__global__ __launch_bounds__(256) void packA_kernel(const float2* __restrict__ xf,
                                                    h16* __restrict__ Afrag,
                                                    int nk, long bStride,
                                                    long jStrideA, long cStride) {
    long gid = (long)blockIdx.x * blockDim.x + threadIdx.x;
    int ep    = (int)(gid & 31);          // 0..31 (re:0-15, im:16-31)
    int lane  = (int)((gid >> 5) & 31);
    int mtile = (int)((gid >> 10) & 1);
    long rest = gid >> 11;
    int kblk  = (int)(rest % nk);
    int c     = (int)(rest / nk);

    int isIm = ep >> 4;
    int e    = ep & 15;
    int hf   = lane >> 4;
    int b    = mtile * 16 + (lane & 15);
    int k    = kblk * 32 + e + 8 * hf + ((e >= 8) ? 8 : 0);
    int j    = k >> 9;
    int t    = k & 511;
    float2 v = xf[(long)b * bStride + (long)j * jStrideA + (long)c * cStride + t];
    Afrag[gid] = (h16)(isIm ? v.y : v.x);
}

// ---------------------------------------------------------------------------
// Complex GEMM via WMMA f16. head h: Y[h][b][s] = sum_k A[b][k] * M[k][s]
//  - A: pre-packed f16 fragments in global (see packA_kernel), loaded with
//    global_load_b128 directly into WMMA operand registers.
//  - M: fp32 complex mask streamed HBM -> LDS with double-buffered
//    global_load_async_to_lds_b128 (ASYNCcnt pipelining), then converted to
//    f16 fragment order in LDS with wide ds ops.
//  Block = 128 threads = 4 waves; 32(M) x 32(N) tile per block; K step = 32.
//  Complex accumulate: Cr = acc_rr - acc_ii ; Ci = acc_ri_ir  (no NEG mods).
// ---------------------------------------------------------------------------
__global__ __launch_bounds__(128) void centangle_gemm_kernel(
        const h16* __restrict__ Afrag,
        const float2* __restrict__ mask, float2* __restrict__ Y,
        int K, int CI, int JC, long jStride) {
    const int h  = blockIdx.z;
    const int i  = h / CI;
    const int c  = h - i * CI;
    const long maskBase = ((long)i * JC + c) * (long)SS * SS;
    const int  n0   = blockIdx.x * 32;
    const int  tid  = threadIdx.x;
    const int  lane = tid & 31;
    const int  wave = tid >> 5;
    const int  mTile = wave & 1;
    const int  nTile = wave >> 1;
    const int  nk = K >> 5;

    __shared__ __align__(16) float2 sRaw[2][32][32];   // raw f32 tiles (16 KB)
    __shared__ __align__(16) h16 sBr[2][32][16];       // B frag [nTile][lane][e]
    __shared__ __align__(16) h16 sBi[2][32][16];

    // conversion role of this thread: column + 8-row k block
    const int colT  = tid & 31;
    const int kbT   = tid >> 5;                 // 0..3  -> k rows kbT*8..+7
    const int laneT = (colT & 15) + 16 * (kbT & 1);
    const int eT    = (kbT >> 1) * 8;
    const int ntT   = colT >> 4;

    v8f acc_rr = {};
    v8f acc_ii = {};
    v8f acc_i  = {};

    // A fragment base: [c][kblk][mTile][lane][32]
    const long aStep = 2 * 32 * 32;
    const long aBase = (((long)c * nk) * 2 + mTile) * (32 * 32) + (long)lane * 32;

    // async-stage one 32x32 float2 tile (8 KB): 4 x b128 per thread
    auto issueAsync = [&](int t, int buf) {
        #pragma unroll
        for (int q = 0; q < 4; ++q) {
            int pairIdx = tid + q * 128;        // 512 x 16B transfers
            int krow = pairIdx >> 4;
            int col  = (pairIdx & 15) * 2;
            int kg = t * 32 + krow;
            int j  = kg >> 9;
            int tt = kg & 511;
            const float2* gp = &mask[maskBase + (long)j * jStride + (long)tt * SS + (n0 + col)];
            unsigned ldsOff = (unsigned)(size_t)(&sRaw[buf][krow][col]);
            asm volatile("global_load_async_to_lds_b128 %0, %1, off"
                         :: "v"(ldsOff), "v"((unsigned long long)(size_t)gp)
                         : "memory");
        }
    };

    issueAsync(0, 0);

    for (int t = 0; t < nk; ++t) {
        const int buf = t & 1;
        if (t + 1 < nk) {
            issueAsync(t + 1, buf ^ 1);
            asm volatile("s_wait_asynccnt 0x4" ::: "memory");  // tile t landed
        } else {
            asm volatile("s_wait_asynccnt 0x0" ::: "memory");
        }
        __syncthreads();

        // convert raw f32 tile -> f16 fragment order (wide LDS ops)
        v8h pr, pi;
        #pragma unroll
        for (int q = 0; q < 8; ++q) {
            float2 v = sRaw[buf][kbT * 8 + q][colT];
            pr[q] = (h16)v.x;
            pi[q] = (h16)v.y;
        }
        *(v8h*)&sBr[ntT][laneT][eT] = pr;
        *(v8h*)&sBi[ntT][laneT][eT] = pi;
        __syncthreads();

        // operand fragments: A straight from global, B from LDS (contiguous)
        const h16* ap = Afrag + aBase + (long)t * aStep;
        v16h ar = *(const v16h*)(ap);
        v16h ai = *(const v16h*)(ap + 16);
        v16h br = *(const v16h*)(&sBr[nTile][lane][0]);
        v16h bi = *(const v16h*)(&sBi[nTile][lane][0]);
        if (t + 1 < nk)
            __builtin_prefetch(Afrag + aBase + (long)(t + 1) * aStep, 0, 1);

        acc_rr = __builtin_amdgcn_wmma_f32_16x16x32_f16(false, ar, false, br, (short)0, acc_rr, false, false);
        acc_ii = __builtin_amdgcn_wmma_f32_16x16x32_f16(false, ai, false, bi, (short)0, acc_ii, false, false);
        acc_i  = __builtin_amdgcn_wmma_f32_16x16x32_f16(false, ar, false, bi, (short)0, acc_i,  false, false);
        acc_i  = __builtin_amdgcn_wmma_f32_16x16x32_f16(false, ai, false, br, (short)0, acc_i,  false, false);
    }

    // store C tile: VGPR r -> row = r + 8*hf (C/D layout), col = lane&15
    const int hf   = lane >> 4;
    const int scol = n0 + nTile * 16 + (lane & 15);
    #pragma unroll
    for (int rIdx = 0; rIdx < 8; ++rIdx) {
        int brow = mTile * 16 + rIdx + 8 * hf;
        Y[((long)h * 32 + brow) * SS + scol] =
            make_float2(acc_rr[rIdx] - acc_ii[rIdx], acc_i[rIdx]);
    }
}

// prod[b,i,c,s] = xf[b,i,c,s] * Yp[(i*8+c)][b][s]
__global__ __launch_bounds__(256) void prod_p_kernel(const float2* __restrict__ xf,
                                                     const float2* __restrict__ Yp,
                                                     float2* __restrict__ prod) {
    long gid = (long)blockIdx.x * blockDim.x + threadIdx.x;  // B*4*C*S
    int s = (int)(gid & 511);
    int c = (int)((gid >> 9) & 7);
    int i = (int)((gid >> 12) & 3);
    int b = (int)(gid >> 14);
    float2 m = Yp[(((long)i * 8 + c) * 32 + b) * SS + s];
    prod[gid] = cmul(xf[gid], m);
}

// tmp2[b,i,s] = xf2[b,i,s] * Yw[i][b][s]
__global__ __launch_bounds__(256) void prod_w_kernel(const float2* __restrict__ xf2,
                                                     const float2* __restrict__ Yw,
                                                     float2* __restrict__ tmp2) {
    long gid = (long)blockIdx.x * blockDim.x + threadIdx.x;  // B*8*S
    int s = (int)(gid & 511);
    int i = (int)((gid >> 9) & 7);
    int b = (int)(gid >> 12);
    float2 m = Yw[((long)i * 32 + b) * SS + s];
    tmp2[gid] = cmul(xf2[gid], m);
}

// esum[b,c,s] = sum_i parietal[b,i,c,s]
__global__ __launch_bounds__(256) void esum_kernel(const float2* __restrict__ par,
                                                   float2* __restrict__ esum) {
    long gid = (long)blockIdx.x * blockDim.x + threadIdx.x;  // B*C*S
    int s = (int)(gid & 511);
    int c = (int)((gid >> 9) & 7);
    int b = (int)(gid >> 12);
    float2 acc = make_float2(0.0f, 0.0f);
    #pragma unroll
    for (int i = 0; i < 4; ++i)
        acc = cadd(acc, par[(((long)b * 4 + i) * CC + c) * SS + s]);
    esum[gid] = acc;
}

// ssum[b,s] = sum_i sup[b,i,s]
__global__ __launch_bounds__(256) void ssum_kernel(const float2* __restrict__ sup,
                                                   float2* __restrict__ ssum) {
    long gid = (long)blockIdx.x * blockDim.x + threadIdx.x;  // B*S
    int s = (int)(gid & 511);
    int b = (int)(gid >> 9);
    float2 acc = make_float2(0.0f, 0.0f);
    #pragma unroll
    for (int i = 0; i < 8; ++i)
        acc = cadd(acc, sup[((long)b * 8 + i) * SS + s]);
    ssum[gid] = acc;
}

// complex softmax over K axis of [B,K,S]; optional complex gain
__global__ __launch_bounds__(256) void isoftmax_kernel(const float2* __restrict__ in,
                                                       float2* __restrict__ out,
                                                       int Kk,
                                                       const float2* __restrict__ gain,
                                                       int useGain) {
    long gid = (long)blockIdx.x * blockDim.x + threadIdx.x;  // B*S
    int s = (int)(gid % SS);
    int b = (int)(gid / SS);
    float mag[8], ex[8];
    float mx = -1e30f;
    for (int k = 0; k < Kk; ++k) {
        float2 v = in[((long)b * Kk + k) * SS + s];
        float m = sqrtf(v.x * v.x + v.y * v.y);
        mag[k] = m;
        mx = fmaxf(mx, m);
    }
    float sum = 0.0f;
    for (int k = 0; k < Kk; ++k) {
        ex[k] = __expf(mag[k] - mx);
        sum += ex[k];
    }
    float2 g = make_float2(1.0f, 0.0f);
    if (useGain) g = gain[0];
    for (int k = 0; k < Kk; ++k) {
        float2 v = in[((long)b * Kk + k) * SS + s];
        float sm  = ex[k] / sum;
        float inv = sm / (mag[k] + 1e-12f);
        float2 o = make_float2(v.x * inv, v.y * inv);
        if (useGain) o = cmul(o, g);
        out[((long)b * Kk + k) * SS + s] = o;
    }
}

// ---------------------------------------------------------------------------
// Bicubic grid sample (align_corners=False, border clamp, a=-0.75).
// img[b,c,h,w] = parietal[b, i=h, c, w] (H=4, W=S); re sampled with grid.re,
// im with grid.im; grid = (warpk[b,0,s], warpk[b,1,s]).
// ---------------------------------------------------------------------------
__device__ __forceinline__ void cubic_w(float t, float wv[4]) {
    const float a = -0.75f;
    float ds[4] = {t + 1.0f, t, 1.0f - t, 2.0f - t};
    #pragma unroll
    for (int q = 0; q < 4; ++q) {
        float ad = fabsf(ds[q]);
        float wq;
        if (ad <= 1.0f)      wq = ((a + 2.0f) * ad - (a + 3.0f)) * ad * ad + 1.0f;
        else if (ad < 2.0f)  wq = a * (ad * (ad * (ad - 5.0f) + 8.0f) - 4.0f);
        else                 wq = 0.0f;
        wv[q] = wq;
    }
}

__device__ __forceinline__ float sample_bc(const float2* __restrict__ par,
                                           int b, int c, float gxn, float gyn,
                                           int comp) {
    const float W = (float)SS, H = 4.0f;
    float gx = ((gxn + 1.0f) * W - 1.0f) * 0.5f;
    float gy = ((gyn + 1.0f) * H - 1.0f) * 0.5f;
    float x0 = floorf(gx), y0 = floorf(gy);
    float wx[4], wy[4];
    cubic_w(gx - x0, wx);
    cubic_w(gy - y0, wy);
    float acc = 0.0f;
    #pragma unroll
    for (int iy = 0; iy < 4; ++iy) {
        float yf = fminf(fmaxf(y0 + (float)(iy - 1), 0.0f), 3.0f);
        int ys = (int)yf;
        float rowacc = 0.0f;
        #pragma unroll
        for (int ix = 0; ix < 4; ++ix) {
            float xf = fminf(fmaxf(x0 + (float)(ix - 1), 0.0f), (float)(SS - 1));
            int xs = (int)xf;
            float2 v = par[(((long)b * 4 + ys) * CC + c) * SS + xs];
            float val = comp ? v.y : v.x;
            rowacc += val * wx[ix];
        }
        acc += rowacc * wy[iy];
    }
    return acc;
}

__global__ __launch_bounds__(256) void bicubic_kernel(const float2* __restrict__ par,
                                                      const float2* __restrict__ warpk,
                                                      float2* __restrict__ warped) {
    long gid = (long)blockIdx.x * blockDim.x + threadIdx.x;  // B*C*S
    int s = (int)(gid & 511);
    int c = (int)((gid >> 9) & 7);
    int b = (int)(gid >> 12);
    float2 g0 = warpk[((long)b * 2 + 0) * SS + s];
    float2 g1 = warpk[((long)b * 2 + 1) * SS + s];
    float re = sample_bc(par, b, c, g0.x, g1.x, 0);
    float im = sample_bc(par, b, c, g0.y, g1.y, 1);
    warped[gid] = make_float2(re, im);
}

// out[i] = a[i]*b[i] elementwise complex, same layout
__global__ __launch_bounds__(256) void cmul_kernel(const float2* __restrict__ a,
                                                   const float2* __restrict__ b,
                                                   float2* __restrict__ o) {
    long gid = (long)blockIdx.x * blockDim.x + threadIdx.x;
    o[gid] = cmul(a[gid], b[gid]);
}

// ---------------------------------------------------------------------------
// Host-side launcher
// ---------------------------------------------------------------------------
static inline size_t alignup(size_t x) { return (x + 255) & ~(size_t)255; }

extern "C" void kernel_launch(void* const* d_in, const int* in_sizes, int n_in,
                              void* d_out, int out_size, void* d_ws, size_t ws_size,
                              hipStream_t stream) {
    (void)in_sizes; (void)n_in; (void)out_size; (void)ws_size;

    const float*  queries = (const float*)d_in[0];   // [B,C,S]
    const float*  states  = (const float*)d_in[1];   // [B,C,S]
    const float2* kw   = (const float2*)d_in[2];     // [5,C,D]
    const float2* kf   = (const float2*)d_in[3];
    const float2* kp   = (const float2*)d_in[4];
    const float2* kr   = (const float2*)d_in[5];     // [5,C]
    const float2* ww   = (const float2*)d_in[6];     // [2,D]
    const float2* wf   = (const float2*)d_in[7];
    const float2* wp   = (const float2*)d_in[8];
    const float2* wrad = (const float2*)d_in[9];     // [2]
    const float2* pmask = (const float2*)d_in[10];   // [4,4,C,S,S]
    const float2* wmask = (const float2*)d_in[11];   // [C,C,1,S,S]
    const float2* gain  = (const float2*)d_in[12];   // [1]

    // ---- workspace carve-up (bytes) ----
    char* ws = (char*)d_ws;
    size_t off = 0;
    auto grab = [&](size_t bytes) { void* p = ws + off; off += alignup(bytes); return p; };

    const size_t BCS  = (size_t)BB * CC * SS;        // 131072
    const size_t B4CS = (size_t)BB * 4 * CC * SS;    // 524288
    const size_t B8S  = (size_t)BB * 8 * SS;         // 131072
    const size_t B2S  = (size_t)BB * 2 * SS;
    const size_t BS   = (size_t)BB * SS;

    float2* integ   = (float2*)grab(BCS  * 8);
    float2* basis   = (float2*)grab(BCS  * 8);
    float2* stacked = (float2*)grab(B4CS * 8);
    float2* xfst    = (float2*)grab(B4CS * 8);
    h16*    AfragP  = (h16*)   grab((size_t)8 * 64 * 2 * 32 * 32 * 2);   // 2 MB
    h16*    AfragW  = (h16*)   grab((size_t)128 * 2 * 32 * 32 * 2);      // 512 KB
    float2* Yp      = (float2*)grab((size_t)32 * 32 * SS * 8);
    float2* prod    = (float2*)grab(B4CS * 8);
    float2* pariet  = (float2*)grab(B4CS * 8);
    float2* esum    = (float2*)grab(BCS * 8);
    float2* xf2     = (float2*)grab(BCS * 8);
    float2* Yw      = (float2*)grab((size_t)8 * 32 * SS * 8);
    float2* tmp2    = (float2*)grab(B8S * 8);
    float2* sup     = (float2*)grab(B8S * 8);
    float2* ssum    = (float2*)grab(BS  * 8);
    float2* wkraw   = (float2*)grab(B2S * 8);
    float2* warpk   = (float2*)grab(B2S * 8);
    float2* ckraw   = (float2*)grab(B8S * 8);
    float2* compk   = (float2*)grab(B8S * 8);
    float2* warped  = (float2*)grab(BCS * 8);
    float2* sig     = (float2*)grab(BCS * 8);
    float2* prodF   = (float2*)grab(BCS * 8);

    const long CS  = (long)CC * SS;
    const long CD  = (long)CC * DD;

    // 1. integ = ifft(states), basis = fft(states)   (real input)
    fft512_kernel<<<BB * CC, 256, 0, stream>>>(integ, states, +1.0f, 1.0f / SS, 1);
    fft512_kernel<<<BB * CC, 256, 0, stream>>>(basis, states, -1.0f, 1.0f, 1);

    // 2. knots -> stacked [B, {ego,basisk,world,integk}, C, S]
    const int knotBlocks8 = (int)((long)BB * CC * SS / 256);
    knot_kernel<<<knotBlocks8, 256, 0, stream>>>(queries, 1, CC, kw + 0 * CD, kf + 0 * CD, kp + 0 * CD, kr + 0 * CC, CC, stacked, 0 * CS, 4 * CS);
    knot_kernel<<<knotBlocks8, 256, 0, stream>>>(basis,   0, CC, kw + 3 * CD, kf + 3 * CD, kp + 3 * CD, kr + 3 * CC, CC, stacked, 1 * CS, 4 * CS);
    knot_kernel<<<knotBlocks8, 256, 0, stream>>>(states,  1, CC, kw + 1 * CD, kf + 1 * CD, kp + 1 * CD, kr + 1 * CC, CC, stacked, 2 * CS, 4 * CS);
    knot_kernel<<<knotBlocks8, 256, 0, stream>>>(integ,   0, CC, kw + 2 * CD, kf + 2 * CD, kp + 2 * CD, kr + 2 * CC, CC, stacked, 3 * CS, 4 * CS);

    // 3. xfst = fft(stacked)
    fft512_kernel<<<BB * 4 * CC, 256, 0, stream>>>(xfst, stacked, -1.0f, 1.0f, 0);

    // 4. pack A fragments, async+WMMA GEMM over pmask (32 heads), finish
    packA_kernel<<<4096, 256, 0, stream>>>(xfst, AfragP, 64,
                                           (long)4 * CC * SS, (long)CC * SS, (long)SS);
    {
        dim3 grid(SS / 32, 1, 32);
        centangle_gemm_kernel<<<grid, 128, 0, stream>>>(AfragP, pmask, Yp,
                                                        2048, 8, 32, (long)CC * SS * SS);
    }
    prod_p_kernel<<<(int)(B4CS / 256), 256, 0, stream>>>(xfst, Yp, prod);
    fft512_kernel<<<BB * 4 * CC, 256, 0, stream>>>(pariet, prod, +1.0f, 1.0f / SS, 0);

    // 5. esum -> xf2 -> async+WMMA GEMM over wmask (8 heads) -> supert -> ssum
    esum_kernel<<<(int)(BCS / 256), 256, 0, stream>>>(pariet, esum);
    fft512_kernel<<<BB * CC, 256, 0, stream>>>(xf2, esum, -1.0f, 1.0f, 0);
    packA_kernel<<<1024, 256, 0, stream>>>(xf2, AfragW, 128,
                                           (long)CC * SS, (long)SS, 0L);
    {
        dim3 grid(SS / 32, 1, 8);
        centangle_gemm_kernel<<<grid, 128, 0, stream>>>(AfragW, wmask, Yw,
                                                        4096, 1, 8, (long)SS * SS);
    }
    prod_w_kernel<<<(int)(B8S / 256), 256, 0, stream>>>(xf2, Yw, tmp2);
    fft512_kernel<<<BB * CC, 256, 0, stream>>>(sup, tmp2, +1.0f, 1.0f / SS, 0);
    ssum_kernel<<<(int)(BS / 256), 256, 0, stream>>>(sup, ssum);

    // 6. warp / compressor knots (collapse Kin=1) + complex softmax
    knot_kernel<<<(int)(B2S / 256), 256, 0, stream>>>(ssum, 0, 1, ww, wf, wp, wrad, 2, wkraw, 0, 2 * (long)SS);
    knot_kernel<<<(int)(B8S / 256), 256, 0, stream>>>(ssum, 0, 1, kw + 4 * CD, kf + 4 * CD, kp + 4 * CD, kr + 4 * CC, CC, ckraw, 0, 8 * (long)SS);
    isoftmax_kernel<<<(int)(BS / 256), 256, 0, stream>>>(wkraw, warpk, 2, gain, 0);
    isoftmax_kernel<<<(int)(BS / 256), 256, 0, stream>>>(ckraw, compk, 8, gain, 1);

    // 7. bicubic warp of parietal (as [B,C,4,S] image), then fft*compk, ifft
    bicubic_kernel<<<(int)(BCS / 256), 256, 0, stream>>>(pariet, warpk, warped);
    fft512_kernel<<<BB * CC, 256, 0, stream>>>(sig, warped, -1.0f, 1.0f, 0);
    cmul_kernel<<<(int)(BCS / 256), 256, 0, stream>>>(sig, compk, prodF);
    fft512_kernel<<<BB * CC, 256, 0, stream>>>((float2*)d_out, prodF, +1.0f, 1.0f / SS, 0);
}
